// MLAttention_37933151158596
// MI455X (gfx1250) — matrix operand order repaired and checked by reading, not compiled, not run on hardware
//
#include <hip/hip_runtime.h>

typedef __attribute__((ext_vector_type(16))) _Float16 v16h;
typedef __attribute__((ext_vector_type(8)))  _Float16 v8h;
typedef __attribute__((ext_vector_type(4)))  _Float16 v4h;
typedef __attribute__((ext_vector_type(8)))  float    v8f;

#define HID     2048
#define NHEAD   16
#define QRANK   768
#define KVRANK  512
#define NOPE    128
#define ROPE_D  64
#define VDIM    128
#define QKD     192      // NOPE + ROPE
#define KVE     (NHEAD*(NOPE+VDIM))   // 4096
#define KVC     (KVRANK+ROPE_D)       // 576
#define SEQ     2048
#define BATCH   2
#define ROWS    (BATCH*SEQ)   // 4096
#define EPS     1e-6f
#define THETA   10000.0f

__device__ __forceinline__ v16h cat8(v8h lo, v8h hi) {
    return __builtin_shufflevector(lo, hi, 0,1,2,3,4,5,6,7,8,9,10,11,12,13,14,15);
}
__device__ __forceinline__ v8h cat4(v4h lo, v4h hi) {
    return __builtin_shufflevector(lo, hi, 0,1,2,3,4,5,6,7);
}
__device__ __forceinline__ v4h cvt4(float4 v) {
    v4h h; h[0] = (_Float16)v.x; h[1] = (_Float16)v.y; h[2] = (_Float16)v.z; h[3] = (_Float16)v.w;
    return h;
}

// ---------------------------------------------------------------------------
// Tiled WMMA GEMM: C[M,N] = A[M,K] * B[K,N]  (f32 in/out, f16 WMMA compute)
// Compile-time K/LDA/LDB/LDC -> immediate-offset addressing, clause-able loads.
// 256 threads = 8 waves; workgroup tile 128x64; K-step 32.
// Each wave: 32x32 register tile = 4 WMMAs/K-step with 2x A/B fragment reuse.
// As : row-major 128x32 (A-fragment = two contiguous b128 runs per lane)
// BsT: transposed 64(n) x 32(k)  (B-fragment = contiguous b128 runs per lane)
// ---------------------------------------------------------------------------
template <int K, int LDA, int LDB, int LDC>
__global__ __launch_bounds__(256) void gemm_wmma_f16(
    const float* __restrict__ A, const float* __restrict__ B, float* __restrict__ C)
{
    __shared__ _Float16 As[128 * 32];   // 8 KB
    __shared__ _Float16 BsT[64 * 32];   // 4 KB

    const int tid  = threadIdx.x;
    const int lane = tid & 31;
    const int wave = tid >> 5;
    const int g    = lane >> 4;
    const int l    = lane & 15;
    const int tM   = blockIdx.y * 128;
    const int tN   = blockIdx.x * 64;
    const int msub = wave & 3;          // rows msub*32 + {0,16}
    const int npair= wave >> 2;         // cols npair*32 + {0,16}

    // staging coordinates
    const int ar   = tid >> 1;          // A row 0..127
    const int akq2 = (tid & 1) * 2;     // A k-octet base (0 or 2)
    const int bc   = tid & 63;          // B column 0..63
    const int brq  = tid >> 6;          // B k-octet 0..3

    const float* pa_base = &A[(size_t)(tM + ar) * LDA];
    const float* pb_base = &B[(size_t)(brq * 8) * LDB + tN + bc];

    v8f c00 = {}, c01 = {}, c10 = {}, c11 = {};

    for (int k0 = 0; k0 < K; k0 += 32) {
        // --- stage A: 4 float4 loads -> two v8h (b128) stores ---
        {
            const float4* pa0 = (const float4*)&pa_base[k0 + akq2 * 8];
            const float4* pa1 = (const float4*)&pa_base[k0 + (akq2 + 1) * 8];
            *(v8h*)&As[ar * 32 + akq2 * 8]       = cat4(cvt4(pa0[0]), cvt4(pa0[1]));
            *(v8h*)&As[ar * 32 + (akq2 + 1) * 8] = cat4(cvt4(pa1[0]), cvt4(pa1[1]));
        }
        // --- stage B transposed: 8 independent (immediate-offset) b32 loads ---
        {
            const float* pb = &pb_base[(size_t)k0 * LDB];
            v8h h;
            #pragma unroll
            for (int j = 0; j < 8; ++j)
                h[j] = (_Float16)pb[(size_t)j * LDB];
            *(v8h*)&BsT[bc * 32 + brq * 8] = h;
        }
        __syncthreads();

        if (k0 + 32 < K) {   // gfx1250 global_prefetch_b8 of next tiles
            __builtin_prefetch(&pa_base[k0 + 32], 0, 1);
            __builtin_prefetch(&pb_base[(size_t)(k0 + 32) * LDB], 0, 1);
        }

        // --- fragments: all b128 LDS loads ---
        const int arow0 = (msub * 32 + l) * 32;
        const int arow1 = (msub * 32 + 16 + l) * 32;
        v16h a0 = cat8(*(const v8h*)&As[arow0 + g * 8],
                       *(const v8h*)&As[arow0 + 16 + g * 8]);
        v16h a1 = cat8(*(const v8h*)&As[arow1 + g * 8],
                       *(const v8h*)&As[arow1 + 16 + g * 8]);
        const int n0 = (npair * 32 + l) * 32;
        const int n1 = (npair * 32 + 16 + l) * 32;
        v16h b0 = cat8(*(const v8h*)&BsT[n0 + g * 8],
                       *(const v8h*)&BsT[n0 + 16 + g * 8]);
        v16h b1 = cat8(*(const v8h*)&BsT[n1 + g * 8],
                       *(const v8h*)&BsT[n1 + 16 + g * 8]);

        c00 = __builtin_amdgcn_wmma_f32_16x16x32_f16(false, a0, false, b0, (short)0, c00, false, false);
        c01 = __builtin_amdgcn_wmma_f32_16x16x32_f16(false, a0, false, b1, (short)0, c01, false, false);
        c10 = __builtin_amdgcn_wmma_f32_16x16x32_f16(false, a1, false, b0, (short)0, c10, false, false);
        c11 = __builtin_amdgcn_wmma_f32_16x16x32_f16(false, a1, false, b1, (short)0, c11, false, false);
        __syncthreads();
    }

    for (int r = 0; r < 8; ++r) {
        int row0 = tM + msub * 32 + r + 8 * g;
        int row1 = row0 + 16;
        int col0 = tN + npair * 32 + l;
        C[(size_t)row0 * LDC + col0]      = c00[r];
        C[(size_t)row0 * LDC + col0 + 16] = c01[r];
        C[(size_t)row1 * LDC + col0]      = c10[r];
        C[(size_t)row1 * LDC + col0 + 16] = c11[r];
    }
}

// ---------------------------------------------------------------------------
// In-place RMS norm over `cols` of each row (row stride ld), float4 traffic
// ---------------------------------------------------------------------------
__global__ __launch_bounds__(256) void rmsnorm_k(float* x, const float* __restrict__ gw,
                                                 int cols, int ld)
{
    __shared__ float red[256];
    float* xr = x + (size_t)blockIdx.x * ld;
    float s = 0.f;
    for (int c = threadIdx.x * 4; c < cols; c += 1024) {
        float4 v = *(const float4*)&xr[c];
        s += v.x * v.x + v.y * v.y + v.z * v.z + v.w * v.w;
    }
    red[threadIdx.x] = s;
    __syncthreads();
    for (int off = 128; off > 0; off >>= 1) {
        if (threadIdx.x < off) red[threadIdx.x] += red[threadIdx.x + off];
        __syncthreads();
    }
    float inv = rsqrtf(red[0] / (float)cols + EPS);
    for (int c = threadIdx.x * 4; c < cols; c += 1024) {
        float4 v = *(const float4*)&xr[c];
        float4 gv = *(const float4*)&gw[c];
        v.x *= inv * gv.x; v.y *= inv * gv.y; v.z *= inv * gv.z; v.w *= inv * gv.w;
        *(float4*)&xr[c] = v;
    }
}

// ---------------------------------------------------------------------------
// RoPE (in place)
// ---------------------------------------------------------------------------
__global__ void rope_q_k(float* q)
{
    const int row = blockIdx.x;
    const int h   = blockIdx.y;
    const int d   = threadIdx.x;              // 0..63
    const int s   = row & (SEQ - 1);
    float* base = q + (size_t)row * (NHEAD * QKD) + h * QKD + NOPE;
    float x  = base[d];
    float xo = base[d < 32 ? d + 32 : d - 32];
    float rot = (d < 32) ? -xo : xo;
    float ang = (float)s * __powf(THETA, -(float)(d & 31) / 32.0f);
    float res = x * __cosf(ang) + rot * __sinf(ang);
    __syncthreads();
    base[d] = res;
}

__global__ void rope_kv_k(float* kv)
{
    const int row = blockIdx.x;
    const int d   = threadIdx.x;              // 0..63
    const int s   = row & (SEQ - 1);
    float* base = kv + (size_t)row * KVC + KVRANK;
    float x  = base[d];
    float xo = base[d < 32 ? d + 32 : d - 32];
    float rot = (d < 32) ? -xo : xo;
    float ang = (float)s * __powf(THETA, -(float)(d & 31) / 32.0f);
    float res = x * __cosf(ang) + rot * __sinf(ang);
    __syncthreads();
    base[d] = res;
}

// ---------------------------------------------------------------------------
// Causal flash attention. 128 threads = 4 waves; wave owns 16 queries; block
// owns 64 queries of one (b,h). 32-key tiles in LDS.
//   Kt : [key][d]   row-major (score B-fragment = contiguous runs)
//   VtT: [n][key]   transposed (PV B-fragment = contiguous runs)
// ---------------------------------------------------------------------------
__global__ __launch_bounds__(128) void mla_attn(
    const float* __restrict__ q, const float* __restrict__ kvexp,
    const float* __restrict__ kvc, float* __restrict__ attn)
{
    __shared__ _Float16 Kt[32 * QKD];        // 12 KB
    __shared__ _Float16 VtT[VDIM * 32];      // 8 KB
    __shared__ _Float16 Ps[4 * 16 * 32];     // 4 KB

    const int tid  = threadIdx.x;
    const int lane = tid & 31;
    const int wave = tid >> 5;
    const int g    = lane >> 4;
    const int l    = lane & 15;
    const int qtile= blockIdx.x;
    const int h    = blockIdx.y;
    const int b    = blockIdx.z;
    const int q0w  = qtile * 64 + wave * 16;
    const float scale = rsqrtf((float)QKD);

    // ---- preload scaled q A-fragments via float4 loads ----
    v16h aq[6];
    {
        const float* qrow = q + (size_t)(b * SEQ + q0w + l) * (NHEAD * QKD) + h * QKD;
        for (int kc = 0; kc < 6; ++kc) {
            const float4* p0 = (const float4*)&qrow[kc * 32 + g * 8];
            const float4* p1 = (const float4*)&qrow[kc * 32 + 16 + g * 8];
            float4 a0 = p0[0], a1 = p0[1], a2 = p1[0], a3 = p1[1];
            float va[16] = { a0.x,a0.y,a0.z,a0.w, a1.x,a1.y,a1.z,a1.w,
                             a2.x,a2.y,a2.z,a2.w, a3.x,a3.y,a3.z,a3.w };
            for (int e = 0; e < 16; ++e) aq[kc][e] = (_Float16)(va[e] * scale);
        }
    }

    v8f acc[8];
    { v8f z = {}; for (int j = 0; j < 8; ++j) acc[j] = z; }
    float mrun[8], lrun[8];
    for (int r = 0; r < 8; ++r) { mrun[r] = -1e30f; lrun[r] = 0.f; }

    // staging coordinates
    const int kkey = tid >> 2;   // 0..31 : key row for K staging
    const int kdq  = tid & 3;    // 0..3  : 48-dim segment
    const int vn   = tid;        // 0..127: vdim column for V staging

    const int t_end = qtile * 64 + 64;
    for (int t0 = 0; t0 < t_end; t0 += 32) {
        // ---- stage K [key][d]: float4 global -> v4h LDS ----
        {
            size_t krow = (size_t)(b * SEQ + t0 + kkey);
            for (int dd = 0; dd < 48; dd += 4) {
                int d = kdq * 48 + dd;
                float4 v4 = (d < NOPE)
                    ? *(const float4*)&kvexp[krow * KVE + h * (NOPE + VDIM) + d]
                    : *(const float4*)&kvc[krow * KVC + KVRANK + (d - NOPE)];
                *(v4h*)&Kt[kkey * QKD + d] = cvt4(v4);
            }
        }
        // ---- stage V transposed [n][key]: coalesced b32 -> v4h LDS ----
        {
            const float* pv = &kvexp[(size_t)(b * SEQ + t0) * KVE + h * (NOPE + VDIM) + NOPE + vn];
            for (int rr = 0; rr < 8; ++rr) {
                v4h hv;
                #pragma unroll
                for (int jj = 0; jj < 4; ++jj)
                    hv[jj] = (_Float16)pv[(size_t)(rr * 4 + jj) * KVE];
                *(v4h*)&VtT[vn * 32 + rr * 4] = hv;
            }
        }
        __syncthreads();

        // ---- scores: 2 subtiles x 6 chained WMMAs, all-b128 fragments ----
        v8f s0 = {}, s1 = {};
        for (int kc = 0; kc < 6; ++kc) {
            const int d0 = kc * 32 + g * 8;
            v16h b0 = cat8(*(const v8h*)&Kt[l * QKD + d0],
                           *(const v8h*)&Kt[l * QKD + d0 + 16]);
            v16h b1 = cat8(*(const v8h*)&Kt[(16 + l) * QKD + d0],
                           *(const v8h*)&Kt[(16 + l) * QKD + d0 + 16]);
            s0 = __builtin_amdgcn_wmma_f32_16x16x32_f16(false, aq[kc], false, b0, (short)0, s0, false, false);
            s1 = __builtin_amdgcn_wmma_f32_16x16x32_f16(false, aq[kc], false, b1, (short)0, s1, false, false);
        }

        // ---- online softmax (row r+8g lives in one 16-lane half) ----
        float p0[8], p1[8];
        for (int r = 0; r < 8; ++r) {
            int   mrow = q0w + r + 8 * g;
            float v0 = ((t0 + l)      <= mrow) ? s0[r] : -1e30f;
            float v1 = ((t0 + 16 + l) <= mrow) ? s1[r] : -1e30f;
            float mx = fmaxf(v0, v1);
            for (int m = 1; m < 16; m <<= 1) mx = fmaxf(mx, __shfl_xor(mx, m, 16));
            float mnew = fmaxf(mrun[r], mx);
            float a0 = __expf(v0 - mnew);
            float a1 = __expf(v1 - mnew);
            float rs = a0 + a1;
            for (int m = 1; m < 16; m <<= 1) rs += __shfl_xor(rs, m, 16);
            float alpha = __expf(mrun[r] - mnew);
            lrun[r] = lrun[r] * alpha + rs;
            mrun[r] = mnew;
            for (int j = 0; j < 8; ++j) { float t = acc[j][r]; acc[j][r] = t * alpha; }
            p0[r] = a0; p1[r] = a1;
        }

        // ---- P: C-layout -> row-major LDS -> A-fragment (b128 reads) ----
        _Float16* Pw = Ps + wave * (16 * 32);
        for (int r = 0; r < 8; ++r) {
            int mrow = r + 8 * g;
            Pw[mrow * 32 + l]      = (_Float16)p0[r];
            Pw[mrow * 32 + 16 + l] = (_Float16)p1[r];
        }
        __asm__ volatile("" ::: "memory");   // LDS is in-order per wave
        v16h ap = cat8(*(const v8h*)&Pw[l * 32 + g * 8],
                       *(const v8h*)&Pw[l * 32 + 16 + g * 8]);

        // ---- O += P x V : 8 WMMAs, contiguous V fragments ----
        for (int j = 0; j < 8; ++j) {
            const int nb = (j * 16 + l) * 32 + g * 8;
            v16h bv = cat8(*(const v8h*)&VtT[nb],
                           *(const v8h*)&VtT[nb + 16]);
            acc[j] = __builtin_amdgcn_wmma_f32_16x16x32_f16(false, ap, false, bv, (short)0, acc[j], false, false);
        }
        __syncthreads();
    }

    // ---- normalize and write attn_out [ROWS, H*VDIM] ----
    for (int r = 0; r < 8; ++r) {
        float invl = 1.0f / lrun[r];
        size_t row = (size_t)(b * SEQ + q0w + r + 8 * g);
        for (int j = 0; j < 8; ++j)
            attn[row * (size_t)(NHEAD * VDIM) + h * VDIM + j * 16 + l] = acc[j][r] * invl;
    }
}

// ---------------------------------------------------------------------------
// Host-side orchestration
// ---------------------------------------------------------------------------
extern "C" void kernel_launch(void* const* d_in, const int* in_sizes, int n_in,
                              void* d_out, int out_size, void* d_ws, size_t ws_size,
                              hipStream_t stream)
{
    const float* hidden = (const float*)d_in[0];
    const float* Wqa    = (const float*)d_in[1];
    const float* g_qa   = (const float*)d_in[2];
    const float* Wqb    = (const float*)d_in[3];
    const float* Wkva   = (const float*)d_in[4];
    const float* g_kva  = (const float*)d_in[5];
    const float* Wkvb   = (const float*)d_in[6];
    const float* Wo     = (const float*)d_in[7];
    float* out = (float*)d_out;

    float* ws    = (float*)d_ws;
    float* qa    = ws;
    float* qbuf  = qa    + (size_t)ROWS * QRANK;
    float* kv    = qbuf  + (size_t)ROWS * NHEAD * QKD;
    float* kvexp = kv    + (size_t)ROWS * KVC;
    float* attn  = kvexp + (size_t)ROWS * KVE;

    dim3 blk(256);

    // 1) q_a = hidden @ Wqa
    gemm_wmma_f16<HID, HID, QRANK, QRANK>
        <<<dim3(QRANK / 64, ROWS / 128), blk, 0, stream>>>(hidden, Wqa, qa);
    rmsnorm_k<<<ROWS, blk, 0, stream>>>(qa, g_qa, QRANK, QRANK);
    // 2) q = q_cn @ Wqb
    gemm_wmma_f16<QRANK, QRANK, NHEAD * QKD, NHEAD * QKD>
        <<<dim3((NHEAD * QKD) / 64, ROWS / 128), blk, 0, stream>>>(qa, Wqb, qbuf);
    // 3) kv = hidden @ Wkva
    gemm_wmma_f16<HID, HID, KVC, KVC>
        <<<dim3(KVC / 64, ROWS / 128), blk, 0, stream>>>(hidden, Wkva, kv);
    rmsnorm_k<<<ROWS, blk, 0, stream>>>(kv, g_kva, KVRANK, KVC);
    // 4) kv_exp = kv_cn @ Wkvb   (A strided by KVC)
    gemm_wmma_f16<KVRANK, KVC, KVE, KVE>
        <<<dim3(KVE / 64, ROWS / 128), blk, 0, stream>>>(kv, Wkvb, kvexp);
    rope_q_k<<<dim3(ROWS, NHEAD), 64, 0, stream>>>(qbuf);
    rope_kv_k<<<ROWS, 64, 0, stream>>>(kv);
    mla_attn<<<dim3(SEQ / 64, NHEAD, BATCH), 128, 0, stream>>>(qbuf, kvexp, kv, attn);
    // 5) out = attn @ Wo
    gemm_wmma_f16<NHEAD * VDIM, NHEAD * VDIM, HID, HID>
        <<<dim3(HID / 64, ROWS / 128), blk, 0, stream>>>(attn, Wo, out);
}
